// AttentionDecoder_41970420416980
// MI455X (gfx1250) — matrix operand order repaired and checked by reading, not compiled, not run on hardware
//
#include <hip/hip_runtime.h>
#include <hip/hip_fp16.h>

typedef __attribute__((ext_vector_type(16))) _Float16 v16h;
typedef __attribute__((ext_vector_type(8)))  float    v8f;

#define U_   256
#define IN_  128
#define B_   128
#define T_   1024
#define OUTD 400   // NUM_MELS * R_FACTOR

// ---------------------------------------------------------------------------
// Hardware tanh (gfx1250 v_tanh_f32 trans op).  Prefer the clang builtin so
// the scheduler knows the TRANS dependency; fall back to inline asm with a
// trailing v_nop (ISA: 1 independent op required after a TRANS before use).
// ---------------------------------------------------------------------------
__device__ __forceinline__ float fast_tanh(float x) {
#if __has_builtin(__builtin_amdgcn_tanhf)
    return __builtin_amdgcn_tanhf(x);
#else
    float r;
    asm volatile("v_tanh_f32 %0, %1\n\tv_nop" : "=v"(r) : "v"(x));
    return r;
#endif
}

// ---------------------------------------------------------------------------
// Convert W1 [U,U] f32 -> f16 row-major (B-matrix source for WMMA)
// ---------------------------------------------------------------------------
__global__ void cvt_w1_kernel(const float* __restrict__ w1, _Float16* __restrict__ w1h) {
    int i = blockIdx.x * blockDim.x + threadIdx.x;
    if (i < U_ * U_) w1h[i] = (_Float16)w1[i];
}

// ---------------------------------------------------------------------------
// Generic GRU cell (PyTorch gate order r,z,n).  One block per batch row,
// 256 threads = one thread per output unit.  Optionally writes x + h_new
// (used for gru2_input / bf_out fusion).
// ---------------------------------------------------------------------------
__global__ void gru_cell_kernel(const float* __restrict__ x, const float* __restrict__ h,
                                const float* __restrict__ Wih, const float* __restrict__ Whh,
                                const float* __restrict__ bih, const float* __restrict__ bhh,
                                float* __restrict__ hout, float* __restrict__ sumout, int xdim) {
    __shared__ float sx[256];
    __shared__ float sh[256];
    int b = blockIdx.x, u = threadIdx.x;
    for (int k = u; k < xdim; k += 256) sx[k] = x[(size_t)b * xdim + k];
    sh[u] = h[(size_t)b * U_ + u];
    __syncthreads();

    float gi[3], gh[3];
#pragma unroll
    for (int g = 0; g < 3; ++g) {
        const float* wr = Wih + (size_t)(g * U_ + u) * xdim;
        float acc = bih[g * U_ + u];
#pragma unroll 8
        for (int k = 0; k < xdim; ++k) acc += sx[k] * wr[k];
        gi[g] = acc;
        const float* hr = Whh + (size_t)(g * U_ + u) * U_;
        float acch = bhh[g * U_ + u];
#pragma unroll 8
        for (int k = 0; k < U_; ++k) acch += sh[k] * hr[k];
        gh[g] = acch;
    }
    float r  = 1.0f / (1.0f + __expf(-(gi[0] + gh[0])));
    float z  = 1.0f / (1.0f + __expf(-(gi[1] + gh[1])));
    float n  = fast_tanh(gi[2] + r * gh[2]);
    float hn = (1.0f - z) * n + z * sh[u];
    hout[(size_t)b * U_ + u] = hn;
    if (sumout) sumout[(size_t)b * U_ + u] = sx[u] + hn;
}

// ---------------------------------------------------------------------------
// query = d_t @ W2^T  (tiny: 128x256x256)
// ---------------------------------------------------------------------------
__global__ void query_kernel(const float* __restrict__ dt, const float* __restrict__ W2,
                             float* __restrict__ query) {
    __shared__ float sd[256];
    int b = blockIdx.x, u = threadIdx.x;
    sd[u] = dt[(size_t)b * U_ + u];
    __syncthreads();
    const float* wr = W2 + (size_t)u * U_;
    float acc = 0.0f;
#pragma unroll 8
    for (int k = 0; k < U_; ++k) acc += sd[k] * wr[k];
    query[(size_t)b * U_ + u] = acc;
}

// ---------------------------------------------------------------------------
// Fused: scores[b,t] = v . tanh( memory[b,t,:] @ W1^T + query[b,:] )
// WMMA f32_16x16x32_f16.  Grid = B*T/128 workgroups of 8 waves; each wave
// owns a 16-row M tile, keeps all 8 A k-fragments in VGPRs, and sweeps
// 16 N-tiles x 8 K-steps = 128 WMMAs.  Keys never touch memory.
// __launch_bounds__(256,1): single-WG occupancy target so the allocator can
// keep the 64-VGPR A panel + B prefetch pipeline fully in registers.
// ---------------------------------------------------------------------------
__global__ void __launch_bounds__(256, 1)
scores_kernel(const float* __restrict__ gmem, const _Float16* __restrict__ w1h,
              const float* __restrict__ query, const float* __restrict__ vvec,
              float* __restrict__ scores) {
    int wave = threadIdx.x >> 5;
    int lane = threadIdx.x & 31;
    int row0 = blockIdx.x * 128 + wave * 16;       // flattened (b,t) row base
    int b     = row0 / T_;
    int tbase = row0 % T_;
    int m  = lane & 15;
    int hi = lane >> 4;                            // lane half select

    // ---- A fragments: 16 rows x 256 K of memory, f32 -> f16 on the fly ----
    const float* arow = gmem + (size_t)(row0 + m) * U_;
    v16h afrag[8];
#pragma unroll
    for (int ks = 0; ks < 8; ++ks) {
        int k0 = ks * 32 + hi * 16;
        float4 c0 = *(const float4*)(arow + k0);
        float4 c1 = *(const float4*)(arow + k0 + 4);
        float4 c2 = *(const float4*)(arow + k0 + 8);
        float4 c3 = *(const float4*)(arow + k0 + 12);
        v16h a;
        a[0]=(_Float16)c0.x;  a[1]=(_Float16)c0.y;  a[2]=(_Float16)c0.z;  a[3]=(_Float16)c0.w;
        a[4]=(_Float16)c1.x;  a[5]=(_Float16)c1.y;  a[6]=(_Float16)c1.z;  a[7]=(_Float16)c1.w;
        a[8]=(_Float16)c2.x;  a[9]=(_Float16)c2.y;  a[10]=(_Float16)c2.z; a[11]=(_Float16)c2.w;
        a[12]=(_Float16)c3.x; a[13]=(_Float16)c3.y; a[14]=(_Float16)c3.z; a[15]=(_Float16)c3.w;
        afrag[ks] = a;
    }

    float partial[8];
#pragma unroll
    for (int j = 0; j < 8; ++j) partial[j] = 0.0f;

    for (int nt = 0; nt < 16; ++nt) {
        int ncol = nt * 16 + m;                    // this lane's output column
        const _Float16* brow = w1h + (size_t)ncol * U_ + hi * 16;
        float qv = query[(size_t)b * U_ + ncol];
        float vv = vvec[ncol];
        v8f acc = {0.f, 0.f, 0.f, 0.f, 0.f, 0.f, 0.f, 0.f};
#pragma unroll
        for (int ks = 0; ks < 8; ++ks) {
            v16h bfrag = *(const v16h*)(brow + ks * 32);
            acc = __builtin_amdgcn_wmma_f32_16x16x32_f16(
                false, afrag[ks], false, bfrag, (short)0, acc, false, false);
        }
#pragma unroll
        for (int j = 0; j < 8; ++j) partial[j] += vv * fast_tanh(acc[j] + qv);
    }

    // reduce over the 16 lanes of each half (columns of the D tile)
#pragma unroll
    for (int j = 0; j < 8; ++j) {
        float p = partial[j];
        p += __shfl_xor(p, 1);
        p += __shfl_xor(p, 2);
        p += __shfl_xor(p, 4);
        p += __shfl_xor(p, 8);
        partial[j] = p;
    }
    if (m == 0) {                                  // lanes 0 (rows 0-7) and 16 (rows 8-15)
        int moff = hi * 8;
#pragma unroll
        for (int j = 0; j < 8; ++j)
            scores[(size_t)b * T_ + tbase + moff + j] = partial[j];
    }
}

// ---------------------------------------------------------------------------
// In-place softmax over T per batch row
// ---------------------------------------------------------------------------
__global__ void softmax_kernel(float* __restrict__ scores) {
    __shared__ float red[256];
    int b = blockIdx.x, tid = threadIdx.x;
    float* s = scores + (size_t)b * T_;
    float mx = -1e30f;
    for (int t = tid; t < T_; t += 256) mx = fmaxf(mx, s[t]);
    red[tid] = mx; __syncthreads();
    for (int st = 128; st > 0; st >>= 1) {
        if (tid < st) red[tid] = fmaxf(red[tid], red[tid + st]);
        __syncthreads();
    }
    mx = red[0]; __syncthreads();
    float sum = 0.0f;
    for (int t = tid; t < T_; t += 256) { float e = __expf(s[t] - mx); s[t] = e; sum += e; }
    red[tid] = sum; __syncthreads();
    for (int st = 128; st > 0; st >>= 1) {
        if (tid < st) red[tid] += red[tid + st];
        __syncthreads();
    }
    float inv = 1.0f / red[0];
    for (int t = tid; t < T_; t += 256) s[t] *= inv;
}

// ---------------------------------------------------------------------------
// context[b,u] = sum_t w[b,t] * memory[b,t,u]   (second, bandwidth-bound pass)
// ---------------------------------------------------------------------------
__global__ void context_kernel(const float* __restrict__ w, const float* __restrict__ gmem,
                               float* __restrict__ ctx) {
    __shared__ float sw[T_];
    int b = blockIdx.x, u = threadIdx.x;
    for (int t = u; t < T_; t += 256) sw[t] = w[(size_t)b * T_ + t];
    __syncthreads();
    const float* mb = gmem + (size_t)b * T_ * U_ + u;
    float acc = 0.0f;
#pragma unroll 4
    for (int t = 0; t < T_; ++t) acc += sw[t] * mb[(size_t)t * U_];
    ctx[(size_t)b * U_ + u] = acc;
}

// ---------------------------------------------------------------------------
// gru1_input = concat(d_t, ctx) @ proj_W^T + proj_b
// ---------------------------------------------------------------------------
__global__ void proj_kernel(const float* __restrict__ dt, const float* __restrict__ ctx,
                            const float* __restrict__ projW, const float* __restrict__ projb,
                            float* __restrict__ out) {
    __shared__ float sd[512];
    int b = blockIdx.x, u = threadIdx.x;
    sd[u]       = dt[(size_t)b * U_ + u];
    sd[256 + u] = ctx[(size_t)b * U_ + u];
    __syncthreads();
    const float* wr = projW + (size_t)u * 512;
    float acc = projb[u];
#pragma unroll 8
    for (int k = 0; k < 512; ++k) acc += sd[k] * wr[k];
    out[(size_t)b * U_ + u] = acc;
}

// ---------------------------------------------------------------------------
// output head: out[b,j] = out_b[j] + bf_out[b] . out_W[j]
// ---------------------------------------------------------------------------
__global__ void out_kernel(const float* __restrict__ bf, const float* __restrict__ outW,
                           const float* __restrict__ outb, float* __restrict__ out) {
    __shared__ float sb[256];
    int b = blockIdx.x, j = threadIdx.x;
    if (j < 256) sb[j] = bf[(size_t)b * U_ + j];
    __syncthreads();
    if (j < OUTD) {
        const float* wr = outW + (size_t)j * U_;
        float acc = outb[j];
#pragma unroll 8
        for (int k = 0; k < U_; ++k) acc += sb[k] * wr[k];
        out[(size_t)b * OUTD + j] = acc;
    }
}

// ---------------------------------------------------------------------------
extern "C" void kernel_launch(void* const* d_in, const int* in_sizes, int n_in,
                              void* d_out, int out_size, void* d_ws, size_t ws_size,
                              hipStream_t stream) {
    const float* decoder_input = (const float*)d_in[0];
    const float* memory        = (const float*)d_in[1];
    const float* attn_hidden   = (const float*)d_in[2];
    const float* gru1_hidden   = (const float*)d_in[3];
    const float* gru2_hidden   = (const float*)d_in[4];
    const float* W1            = (const float*)d_in[5];
    const float* W2            = (const float*)d_in[6];
    const float* vvec          = (const float*)d_in[7];
    const float* attn_Wih      = (const float*)d_in[8];
    const float* attn_Whh      = (const float*)d_in[9];
    const float* attn_bih      = (const float*)d_in[10];
    const float* attn_bhh      = (const float*)d_in[11];
    const float* g1_Wih        = (const float*)d_in[12];
    const float* g1_Whh        = (const float*)d_in[13];
    const float* g1_bih        = (const float*)d_in[14];
    const float* g1_bhh        = (const float*)d_in[15];
    const float* g2_Wih        = (const float*)d_in[16];
    const float* g2_Whh        = (const float*)d_in[17];
    const float* g2_bih        = (const float*)d_in[18];
    const float* g2_bhh        = (const float*)d_in[19];
    const float* proj_W        = (const float*)d_in[20];
    const float* proj_b        = (const float*)d_in[21];
    const float* out_W         = (const float*)d_in[22];
    const float* out_b         = (const float*)d_in[23];

    // output tuple layout: output[B,80,5] , d_t[B,U] , gru1_h[B,U] , gru2_h[B,U]
    float* out = (float*)d_out;
    float* dt  = out + (size_t)B_ * OUTD;     // 51200
    float* g1h = dt  + (size_t)B_ * U_;       // +32768
    float* g2h = g1h + (size_t)B_ * U_;       // +32768

    // workspace carve-up (~1.25 MB)
    char* ws = (char*)d_ws;
    _Float16* w1h  = (_Float16*)ws;                               // 128 KB
    float* scores  = (float*)(ws + 131072);                        // 512 KB
    float* query   = (float*)(ws + 131072 + 524288);               // 128 KB
    float* ctx     = query + (size_t)B_ * U_;
    float* g1in    = ctx   + (size_t)B_ * U_;
    float* g2in    = g1in  + (size_t)B_ * U_;
    float* bfout   = g2in  + (size_t)B_ * U_;

    cvt_w1_kernel<<<256, 256, 0, stream>>>(W1, w1h);

    gru_cell_kernel<<<B_, 256, 0, stream>>>(decoder_input, attn_hidden,
                                            attn_Wih, attn_Whh, attn_bih, attn_bhh,
                                            dt, nullptr, IN_);

    query_kernel<<<B_, 256, 0, stream>>>(dt, W2, query);

    scores_kernel<<<(B_ * T_) / 128, 256, 0, stream>>>(memory, w1h, query, vvec, scores);

    softmax_kernel<<<B_, 256, 0, stream>>>(scores);

    context_kernel<<<B_, 256, 0, stream>>>(scores, memory, ctx);

    proj_kernel<<<B_, 256, 0, stream>>>(dt, ctx, proj_W, proj_b, g1in);

    gru_cell_kernel<<<B_, 256, 0, stream>>>(g1in, gru1_hidden,
                                            g1_Wih, g1_Whh, g1_bih, g1_bhh,
                                            g1h, g2in, U_);

    gru_cell_kernel<<<B_, 256, 0, stream>>>(g2in, gru2_hidden,
                                            g2_Wih, g2_Whh, g2_bih, g2_bhh,
                                            g2h, bfout, U_);

    out_kernel<<<B_, 512, 0, stream>>>(bfout, out_W, out_b, out);
}
